// Net_46359876993286
// MI455X (gfx1250) — matrix operand order repaired and checked by reading, not compiled, not run on hardware
//
#include <hip/hip_runtime.h>

typedef __attribute__((ext_vector_type(8)))  float  v8f;
typedef __attribute__((ext_vector_type(8)))  __bf16 v8bf;
typedef __attribute__((ext_vector_type(16))) __bf16 v16bf;
typedef __attribute__((ext_vector_type(4)))  unsigned int uint4v;
typedef __attribute__((ext_vector_type(4)))  int int4v;
typedef __attribute__((ext_vector_type(8)))  int int8v;

#define TPB 256
#define KT  512                      // K-chunk staged in LDS per block
#define LDS_ROW (KT + 8)             // +16B pad per row (TDM pad_enable) -> no bank conflicts

#if __has_builtin(__builtin_amdgcn_tensor_load_to_lds) && __has_builtin(__builtin_amdgcn_s_wait_tensorcnt)
#define HAVE_TDM 1
#else
#define HAVE_TDM 0
#endif

__device__ __forceinline__ unsigned short f32_to_bf16(float f) {
    unsigned int u = __float_as_uint(f);
    u += 0x7FFFu + ((u >> 16) & 1u);          // round-to-nearest-even
    return (unsigned short)(u >> 16);
}

__device__ __forceinline__ void atomicMaxF(float* addr, float val) {
    int* ai = (int*)addr;
    int old = __float_as_int(*addr);
    while (__int_as_float(old) < val) {
        int assumed = old;
        old = atomicCAS(ai, assumed, __float_as_int(val));
        if (old == assumed) break;
    }
}

// ---------- TDM: async 2D tile load (16 rows x KT bf16) global -> LDS ----------
// D# built per CDNA5 ISA 8.3/8.4: data_size=2B, pad 4 dwords every 256 dwords.
__device__ __forceinline__ void tdm_load_tile(const unsigned short* gsrc, unsigned int lds_off,
                                              int lda_elems) {
#if HAVE_TDM
    unsigned long long ga = (unsigned long long)gsrc;
    uint4v g0;
    g0.x = 1u;                                              // count=1 (valid), no gather
    g0.y = lds_off;                                         // lds_addr (bytes)
    g0.z = (unsigned)(ga & 0xffffffffu);                    // global_addr[31:0]
    g0.w = (unsigned)((ga >> 32) & 0x1ffffffu) | 0x80000000u; // addr[56:32] | type=2
    const unsigned td0 = (unsigned)KT;                      // tensor_dim0 (elements)
    const unsigned td1 = 16u;                               // tensor_dim1 (rows)
    const unsigned long long s0 = (unsigned long long)lda_elems;
    int8v g1;
    g1[0] = (int)(0x00010000u                               // data_size=1 -> 2 bytes
                  | (1u << 20)                              // pad_enable
                  | (7u << 22)                              // pad_interval: every 256 dwords
                  | (3u << 25));                            // pad_amount: 4 dwords (16B)
    g1[1] = (int)((td0 & 0xffffu) << 16);                   // tensor_dim0[15:0]
    g1[2] = (int)((td0 >> 16) | ((td1 & 0xffffu) << 16));   // tensor_dim0[31:16], dim1[15:0]
    g1[3] = (int)((td1 >> 16) | ((unsigned)KT << 16));      // tensor_dim1[31:16], tile_dim0
    g1[4] = 16;                                             // tile_dim1=16, tile_dim2=0
    g1[5] = (int)(s0 & 0xffffffffu);                        // tensor_dim0_stride[31:0]
    g1[6] = (int)((s0 >> 32) & 0xffffu);                    // stride hi, dim1_stride lo = 0
    g1[7] = 0;
    int4v gz = {0, 0, 0, 0};
#if __clang_major__ >= 23
    int8v gz8 = {0, 0, 0, 0, 0, 0, 0, 0};
    __builtin_amdgcn_tensor_load_to_lds(g0, g1, gz, gz, gz8, 0);
#else
    __builtin_amdgcn_tensor_load_to_lds(g0, g1, gz, gz, 0);
#endif
#endif
}

// ---------- elementwise / conversion kernels ----------

__global__ void k_fill(float* __restrict__ p, float v, long long n) {
    long long i = (long long)blockIdx.x * blockDim.x + threadIdx.x;
    if (i < n) p[i] = v;
}

// out[r][c] (R x Cpad, bf16) = in[r][c] (R x C, f32), zero-padded cols (coalesced both sides)
__global__ void k_cvt_bf16_pad(const float* __restrict__ in, unsigned short* __restrict__ out,
                               int R, int C, int Cpad) {
    long long i = (long long)blockIdx.x * blockDim.x + threadIdx.x;
    long long total = (long long)R * Cpad;
    if (i >= total) return;
    int r = (int)(i / Cpad), c = (int)(i % Cpad);
    out[i] = (c < C) ? f32_to_bf16(in[(long long)r * C + c]) : (unsigned short)0;
}

// spectral filter values per node: scl[r] = h_mode(lamb[r])
__global__ void k_filter(const float* __restrict__ lamb, float* __restrict__ scl,
                         int mode, int Nn) {
    int i = blockIdx.x * blockDim.x + threadIdx.x;
    if (i >= Nn) return;
    float lam = lamb[i], hv;
    if (mode == 0) {                            // phi(8*lamb)
        float l = 8.0f * lam;
        if (l == 0.0f) l = 1e-7f;
        float y = 6.283185307179586f * l;
        hv = sinf(y) / y;
    } else {                                    // psi(lamb/4) or psi(lamb/2)
        float l = (mode == 1) ? lam * 0.25f : lam * 0.5f;
        if (l == 0.0f) l = 1e-8f;
        float y = 3.141592653589793f * l;
        hv = sinf(l) / y * (1.0f - cosf(l));
    }
    scl[i] = hv;
}

// LDS-tiled transpose: out[c][r] (C x Rpad, bf16) = scale[r] * (abs?)in[r][c]; zero-padded rows
// grid: ((C+31)/32, (Rpad+31)/32), block (32,8)
__global__ void k_transpose_scale_bf16(const float* __restrict__ in, unsigned short* __restrict__ out,
                                       int R, int C, int Rpad, int do_abs,
                                       const float* __restrict__ scale) {
    __shared__ float tile[32][33];
    const int cb = blockIdx.x * 32, rb = blockIdx.y * 32;
    const int tx = threadIdx.x, ty = threadIdx.y;
#pragma unroll
    for (int j = 0; j < 32; j += 8) {
        int r = rb + ty + j, c = cb + tx;
        float v = 0.0f;
        if (r < R && c < C) {
            v = in[(size_t)r * C + c];
            if (do_abs) v = fabsf(v);
            if (scale)  v *= scale[r];
        }
        tile[ty + j][tx] = v;
    }
    __syncthreads();
#pragma unroll
    for (int j = 0; j < 32; j += 8) {
        int c = cb + ty + j, r = rb + tx;
        if (c < C && r < Rpad)
            out[(size_t)c * Rpad + r] = f32_to_bf16(tile[tx][ty + j]);
    }
}

// ---------- WMMA GEMM, direct global loads (small K) ----------
// C[16 rows, ncols] = A[M x K](bf16, ld=lda) @ Bt[ncols x K]^T (+bias)
// grid.x = M/16, blockDim = 32*(ncols/16)
__global__ void k_wmma_gemm(const unsigned short* __restrict__ A,
                            const unsigned short* __restrict__ Bt,
                            float* __restrict__ C,
                            int Kp, int lda, int ldc,
                            const float* __restrict__ bias) {
    const int wave = threadIdx.x >> 5;
    const int lane = threadIdx.x & 31;
    const int row0 = blockIdx.x << 4;
    const int col0 = wave << 4;
    const int rA  = row0 + (lane & 15);
    const int kbA = (lane & 16) ? 8 : 0;
    const int cB  = col0 + (lane & 15);
    const int kbB = (lane & 16) ? 8 : 0;

    const unsigned short* pA = A  + (size_t)rA * lda + kbA;
    const unsigned short* pB = Bt + (size_t)cB * lda + kbB;

    v8f acc = {};
    for (int k = 0; k < Kp; k += 32) {
        v8bf a0 = *(const v8bf*)(pA);
        v8bf a1 = *(const v8bf*)(pA + 16);
        v8bf b0 = *(const v8bf*)(pB);
        v8bf b1 = *(const v8bf*)(pB + 16);
        v16bf av = __builtin_shufflevector(a0, a1, 0,1,2,3,4,5,6,7,8,9,10,11,12,13,14,15);
        v16bf bv = __builtin_shufflevector(b0, b1, 0,1,2,3,4,5,6,7,8,9,10,11,12,13,14,15);
        acc = __builtin_amdgcn_wmma_f32_16x16x32_bf16(false, av, false, bv,
                                                      (short)0, acc, false, false);
        pA += 32; pB += 32;
    }

    const int srow = row0 + ((lane & 16) ? 8 : 0);
    const int scol = col0 + (lane & 15);
    const float bv = bias ? bias[scol] : 0.0f;
#pragma unroll
    for (int v = 0; v < 8; ++v)
        C[(size_t)(srow + v) * ldc + scol] = acc[v] + bv;
}

// ---------- WMMA GEMM, A staged in LDS via TDM double-buffer (big K) ----------
// C[16 rows, 128 cols] = A[M x Kp](bf16, ld=Kp) @ Bt[128 x Kp]^T ; Kp % KT == 0
// grid.x = M/16, blockDim = 256 (8 waves)
__global__ __launch_bounds__(256) void k_wmma_gemm_lds(const unsigned short* __restrict__ A,
                                                       const unsigned short* __restrict__ Bt,
                                                       float* __restrict__ C,
                                                       int Kp, int ldc) {
    __shared__ unsigned short Abuf[2][16 * LDS_ROW];   // 2 x 16.25KB
    const int wave = threadIdx.x >> 5;
    const int lane = threadIdx.x & 31;
    const int row0 = blockIdx.x << 4;
    const int col0 = wave << 4;
    const int kbA = (lane & 16) ? 8 : 0;
    const int cB  = col0 + (lane & 15);
    const int kbB = (lane & 16) ? 8 : 0;
    const int nchunk = Kp / KT;

    auto load_chunk = [&](int buf, int k0) {
#if HAVE_TDM
        if (wave == 0)
            tdm_load_tile(A + (size_t)row0 * Kp + k0,
                          (unsigned)(unsigned long long)&Abuf[buf][0], Kp);
#else
        for (int idx = threadIdx.x; idx < 16 * (KT / 8); idx += 256) {
            int r = idx / (KT / 8), c8 = (idx % (KT / 8)) * 8;
            *(v8bf*)&Abuf[buf][r * LDS_ROW + c8] =
                *(const v8bf*)(A + (size_t)(row0 + r) * Kp + k0 + c8);
        }
#endif
    };
    auto wait_chunk = [&]() {
#if HAVE_TDM
        if (wave == 0) __builtin_amdgcn_s_wait_tensorcnt(0);
#endif
        __syncthreads();
    };

    v8f acc = {};
    load_chunk(0, 0);
    wait_chunk();
    int cur = 0;
    for (int ch = 0; ch < nchunk; ++ch) {
        const int k0 = ch * KT;
        if (ch + 1 < nchunk) load_chunk(cur ^ 1, k0 + KT);

        const unsigned short* la = &Abuf[cur][(lane & 15) * LDS_ROW + kbA];
        const unsigned short* pB = Bt + (size_t)cB * Kp + k0 + kbB;
#pragma unroll 4
        for (int kk = 0; kk < KT; kk += 32) {
            v8bf a0 = *(const v8bf*)(la);
            v8bf a1 = *(const v8bf*)(la + 16);
            v8bf b0 = *(const v8bf*)(pB);
            v8bf b1 = *(const v8bf*)(pB + 16);
            v16bf av = __builtin_shufflevector(a0, a1, 0,1,2,3,4,5,6,7,8,9,10,11,12,13,14,15);
            v16bf bv = __builtin_shufflevector(b0, b1, 0,1,2,3,4,5,6,7,8,9,10,11,12,13,14,15);
            acc = __builtin_amdgcn_wmma_f32_16x16x32_bf16(false, av, false, bv,
                                                          (short)0, acc, false, false);
            la += 32; pB += 32;
            __builtin_prefetch(pB + 224, 0, 1);
        }
        wait_chunk();
        cur ^= 1;
    }

    const int srow = row0 + ((lane & 16) ? 8 : 0);
    const int scol = col0 + (lane & 15);
#pragma unroll
    for (int v = 0; v < 8; ++v)
        C[(size_t)(srow + v) * ldc + scol] = acc[v];
}

// ---------- edge / graph kernels ----------

__global__ void k_edge_gather_add(const float* __restrict__ rep, const long long* __restrict__ src,
                                  const long long* __restrict__ dst, float* __restrict__ agg,
                                  long long E_, int Ccols) {
    long long i = (long long)blockIdx.x * blockDim.x + threadIdx.x;
    long long total = E_ * Ccols;
    if (i >= total) return;
    long long e = i / Ccols; int c = (int)(i % Ccols);
    atomicAdd(&agg[dst[e] * Ccols + c], rep[src[e] * Ccols + c]);
}

__global__ void k_att(const float* __restrict__ hgat, const float* __restrict__ att_src,
                      const float* __restrict__ att_dst, float* __restrict__ asrc,
                      float* __restrict__ adst, int Nn) {
    int i = blockIdx.x * blockDim.x + threadIdx.x;
    if (i >= Nn * 2) return;
    int n = i >> 1, hh = i & 1;
    float s1 = 0.0f, s2 = 0.0f;
#pragma unroll
    for (int c = 0; c < 16; ++c) {
        float hv = hgat[n * 32 + hh * 16 + c];
        s1 += hv * att_src[hh * 16 + c];
        s2 += hv * att_dst[hh * 16 + c];
    }
    asrc[i] = s1; adst[i] = s2;
}

__device__ __forceinline__ void edge_sd(long long e, const long long* src, const long long* dst,
                                        long long E_, long long& s, long long& d) {
    if (e < E_) { s = src[e]; d = dst[e]; } else { s = d = e - E_; }
}

__global__ void k_edge_max(const float* __restrict__ asrc, const float* __restrict__ adst,
                           const long long* __restrict__ src, const long long* __restrict__ dst,
                           float* __restrict__ m, long long E_, int Nn) {
    long long i = (long long)blockIdx.x * blockDim.x + threadIdx.x;
    if (i >= (E_ + Nn) * 2) return;
    long long e = i >> 1; int hh = (int)(i & 1);
    long long s, d; edge_sd(e, src, dst, E_, s, d);
    float v = asrc[s * 2 + hh] + adst[d * 2 + hh];
    v = v > 0.0f ? v : 0.2f * v;
    atomicMaxF(&m[d * 2 + hh], v);
}

__global__ void k_edge_denom(const float* __restrict__ asrc, const float* __restrict__ adst,
                             const long long* __restrict__ src, const long long* __restrict__ dst,
                             const float* __restrict__ m, float* __restrict__ den,
                             long long E_, int Nn) {
    long long i = (long long)blockIdx.x * blockDim.x + threadIdx.x;
    if (i >= (E_ + Nn) * 2) return;
    long long e = i >> 1; int hh = (int)(i & 1);
    long long s, d; edge_sd(e, src, dst, E_, s, d);
    float v = asrc[s * 2 + hh] + adst[d * 2 + hh];
    v = v > 0.0f ? v : 0.2f * v;
    atomicAdd(&den[d * 2 + hh], expf(v - m[d * 2 + hh]));
}

__global__ void k_edge_accum(const float* __restrict__ asrc, const float* __restrict__ adst,
                             const float* __restrict__ m, const float* __restrict__ den,
                             const float* __restrict__ hgat,
                             const long long* __restrict__ src, const long long* __restrict__ dst,
                             float* __restrict__ gout, long long E_, int Nn) {
    long long i = (long long)blockIdx.x * blockDim.x + threadIdx.x;
    if (i >= (E_ + Nn) * 2) return;
    long long e = i >> 1; int hh = (int)(i & 1);
    long long s, d; edge_sd(e, src, dst, E_, s, d);
    float v = asrc[s * 2 + hh] + adst[d * 2 + hh];
    v = v > 0.0f ? v : 0.2f * v;
    float alpha = expf(v - m[d * 2 + hh]) / den[d * 2 + hh];
    const float* hs = hgat + s * 32 + hh * 16;
    float* go = gout + d * 32 + hh * 16;
#pragma unroll
    for (int c = 0; c < 16; ++c) atomicAdd(&go[c], alpha * hs[c]);
}

__global__ void k_mlp(const float* __restrict__ gout, const float* __restrict__ gbias,
                      const float* __restrict__ mW, const float* __restrict__ mb,
                      float* __restrict__ g2, int Nn) {
    int i = blockIdx.x * blockDim.x + threadIdx.x;
    if (i >= Nn * 16) return;
    int n = i >> 4, j = i & 15;
    float sum = mb[j];
#pragma unroll
    for (int k = 0; k < 32; ++k) {
        float t = gout[n * 32 + k] + gbias[k];
        t = t > 0.0f ? t : (expf(t) - 1.0f);   // elu
        sum += t * mW[k * 16 + j];
    }
    g2[i] = sum;
}

__global__ void k_edge_out(const float* __restrict__ g2, const long long* __restrict__ src,
                           const long long* __restrict__ dst, float* __restrict__ oac, long long E_) {
    long long i = (long long)blockIdx.x * blockDim.x + threadIdx.x;
    if (i >= E_ * 16) return;
    long long e = i >> 4; int c = (int)(i & 15);
    atomicAdd(&oac[dst[e] * 16 + c], g2[src[e] * 16 + c]);
}

__global__ void k_lsm(const float* __restrict__ oac, float* __restrict__ out, int Nn) {
    int i = blockIdx.x * blockDim.x + threadIdx.x;
    if (i >= Nn) return;
    const float* xr = oac + (long long)i * 16;
    float mx = xr[0];
#pragma unroll
    for (int c = 1; c < 16; ++c) mx = fmaxf(mx, xr[c]);
    float s = 0.0f;
#pragma unroll
    for (int c = 0; c < 16; ++c) s += expf(xr[c] - mx);
    float lse = logf(s) + mx;
    float* orow = out + (long long)i * 16;
#pragma unroll
    for (int c = 0; c < 16; ++c) orow[c] = xr[c] - lse;
}

// ---------- host orchestration ----------

extern "C" void kernel_launch(void* const* d_in, const int* in_sizes, int n_in,
                              void* d_out, int out_size, void* d_ws, size_t ws_size,
                              hipStream_t stream) {
    const float*     x     = (const float*)d_in[0];
    const long long* ei    = (const long long*)d_in[1];
    const float*     lamb  = (const float*)d_in[2];
    const float*     V     = (const float*)d_in[3];
    const float*     W1    = (const float*)d_in[4];
    const float*     b1    = (const float*)d_in[5];
    const float*     gat_W = (const float*)d_in[6];
    const float*     att_s = (const float*)d_in[7];
    const float*     att_d = (const float*)d_in[8];
    const float*     gat_b = (const float*)d_in[9];
    const float*     mlp_W = (const float*)d_in[10];
    const float*     mlp_b = (const float*)d_in[11];
    float* out = (float*)d_out;

    const int N = 10000;
    const long long E_ = 320000;
    const int Kp = 10240;                       // 10000 padded to multiple of KT(=512)

    const long long* src = ei;
    const long long* dst = ei + E_;

    char* w = (char*)d_ws;
    auto alloc = [&](size_t bytes) -> void* {
        void* p = (void*)w;
        w += (bytes + 255) & ~(size_t)255;
        return p;
    };
    unsigned short* Vb   = (unsigned short*)alloc((size_t)N * Kp * 2);
    unsigned short* Vtb  = (unsigned short*)alloc((size_t)N * Kp * 2);
    unsigned short* xtb  = (unsigned short*)alloc((size_t)128 * Kp * 2);
    unsigned short* ytb  = (unsigned short*)alloc((size_t)128 * Kp * 2);
    unsigned short* xb   = (unsigned short*)alloc((size_t)N * 128 * 2);
    unsigned short* W1tb = (unsigned short*)alloc((size_t)128 * 128 * 2);
    unsigned short* gWtb = (unsigned short*)alloc((size_t)32 * 896 * 2);
    unsigned short* aggb = (unsigned short*)alloc((size_t)N * 896 * 2);
    float* h    = (float*)alloc((size_t)N * 128 * 4);
    float* VtX  = (float*)alloc((size_t)N * 128 * 4);
    float* scl  = (float*)alloc((size_t)N * 4);
    float* t[6];
    for (int i = 0; i < 6; ++i) t[i] = (float*)alloc((size_t)N * 128 * 4);
    float* rep  = (float*)alloc((size_t)N * 896 * 4);
    float* agg  = (float*)alloc((size_t)N * 896 * 4);
    float* hgat = (float*)alloc((size_t)N * 32 * 4);
    float* asrc = (float*)alloc((size_t)N * 2 * 4);
    float* adst = (float*)alloc((size_t)N * 2 * 4);
    float* mrow = (float*)alloc((size_t)N * 2 * 4);
    float* den  = (float*)alloc((size_t)N * 2 * 4);
    float* gout = (float*)alloc((size_t)N * 32 * 4);
    float* g2   = (float*)alloc((size_t)N * 16 * 4);
    float* oac  = (float*)alloc((size_t)N * 16 * 4);

    auto blocks = [](long long n) { return (unsigned)((n + TPB - 1) / TPB); };
    const dim3 tb(32, 8);
    auto tgrid = [](int C, int Rpad) { return dim3((C + 31) / 32, (Rpad + 31) / 32); };

    // one-time bf16 conversions (tiled transposes go through LDS)
    k_cvt_bf16_pad<<<blocks((long long)N * Kp), TPB, 0, stream>>>(V, Vb, N, N, Kp);
    k_transpose_scale_bf16<<<tgrid(N, Kp), tb, 0, stream>>>(V, Vtb, N, N, Kp, 0, nullptr);
    k_cvt_bf16_pad<<<blocks((long long)N * 128), TPB, 0, stream>>>(x, xb, N, 128, 128);
    k_transpose_scale_bf16<<<tgrid(128, 128), tb, 0, stream>>>(W1, W1tb, 128, 128, 128, 0, nullptr);
    k_transpose_scale_bf16<<<tgrid(32, 896), tb, 0, stream>>>(gat_W, gWtb, 896, 32, 896, 0, nullptr);

    // h = x @ W1 + b1
    k_wmma_gemm<<<N / 16, 256, 0, stream>>>(xb, W1tb, h, 128, 128, 128, b1);

    // _prop3: Vt_x = V^T @ inp ; o_i = V @ (h_i(lamb) * Vt_x)
    auto prop3 = [&](const float* inp, int do_abs, float* o0, int ldc0, float* o1, float* o2) {
        k_transpose_scale_bf16<<<tgrid(128, Kp), tb, 0, stream>>>(inp, xtb, N, 128, Kp, do_abs, nullptr);
        k_wmma_gemm_lds<<<N / 16, 256, 0, stream>>>(Vtb, xtb, VtX, Kp, 128);
        float* outs[3] = {o0, o1, o2};
        int    ldcs[3] = {ldc0, 128, 128};
        for (int mode = 0; mode < 3; ++mode) {
            if (!outs[mode]) continue;
            k_filter<<<blocks(N), TPB, 0, stream>>>(lamb, scl, mode, N);
            k_transpose_scale_bf16<<<tgrid(128, Kp), tb, 0, stream>>>(VtX, ytb, N, 128, Kp, 0, scl);
            k_wmma_gemm_lds<<<N / 16, 256, 0, stream>>>(Vb, ytb, outs[mode], Kp, ldcs[mode]);
        }
    };

    // _get_rep tree (only outputs that are actually consumed)
    prop3(h,    0, rep + 0,   896, t[0], t[1]);
    prop3(t[0], 1, rep + 128, 896, t[2], t[3]);
    prop3(t[1], 1, rep + 256, 896, t[4], t[5]);
    prop3(t[2], 1, rep + 384, 896, nullptr, nullptr);
    prop3(t[3], 1, rep + 512, 896, nullptr, nullptr);
    prop3(t[4], 1, rep + 640, 896, nullptr, nullptr);
    prop3(t[5], 1, rep + 768, 896, nullptr, nullptr);

    // agg = segment_sum(rep[src], dst)
    k_fill<<<blocks((long long)N * 896), TPB, 0, stream>>>(agg, 0.0f, (long long)N * 896);
    k_edge_gather_add<<<blocks(E_ * 896), TPB, 0, stream>>>(rep, src, dst, agg, E_, 896);

    // GAT projection: hgat = agg @ gat_W   (10000 x 896 x 32)
    k_cvt_bf16_pad<<<blocks((long long)N * 896), TPB, 0, stream>>>(agg, aggb, N, 896, 896);
    k_wmma_gemm<<<N / 16, 64, 0, stream>>>(aggb, gWtb, hgat, 896, 896, 32, nullptr);

    // attention + edge softmax (with self loops)
    k_att<<<blocks((long long)N * 2), TPB, 0, stream>>>(hgat, att_s, att_d, asrc, adst, N);
    k_fill<<<blocks((long long)N * 2), TPB, 0, stream>>>(mrow, -1e30f, (long long)N * 2);
    k_fill<<<blocks((long long)N * 2), TPB, 0, stream>>>(den, 0.0f, (long long)N * 2);
    k_fill<<<blocks((long long)N * 32), TPB, 0, stream>>>(gout, 0.0f, (long long)N * 32);
    k_edge_max  <<<blocks((E_ + N) * 2), TPB, 0, stream>>>(asrc, adst, src, dst, mrow, E_, N);
    k_edge_denom<<<blocks((E_ + N) * 2), TPB, 0, stream>>>(asrc, adst, src, dst, mrow, den, E_, N);
    k_edge_accum<<<blocks((E_ + N) * 2), TPB, 0, stream>>>(asrc, adst, mrow, den, hgat, src, dst,
                                                           gout, E_, N);

    // g2 = elu(gout + gat_bias) @ mlp_W + mlp_b ; out = log_softmax(segment_sum(g2[src], dst))
    k_mlp<<<blocks((long long)N * 16), TPB, 0, stream>>>(gout, gat_b, mlp_W, mlp_b, g2, N);
    k_fill<<<blocks((long long)N * 16), TPB, 0, stream>>>(oac, 0.0f, (long long)N * 16);
    k_edge_out<<<blocks(E_ * 16), TPB, 0, stream>>>(g2, src, dst, oac, E_);
    k_lsm<<<blocks((long long)N), TPB, 0, stream>>>(oac, out, N);
}
